// LinearAttention_50208167690717
// MI455X (gfx1250) — compile-verified, matrix operand order checked by reading, not run on hardware
//
#include <hip/hip_runtime.h>
#include <hip/hip_bf16.h>
#include <math.h>

// ---------------- problem constants ----------------
constexpr int nB  = 2;
constexpr int nS  = 8192;
constexpr int nD  = 1024;
constexpr int nH  = 16;
constexpr int nHD = 64;
constexpr int nFD = 256;
constexpr int nBH = nB * nH;      // 32
constexpr int NCH = 16;           // S-reduction slices for kv/ksum partials
constexpr int SC  = nS / NCH;     // 512

// ---------------- vector types ----------------
typedef _Float16 v16h __attribute__((ext_vector_type(16)));
typedef _Float16 v8h  __attribute__((ext_vector_type(8)));
typedef float    v8f  __attribute__((ext_vector_type(8)));
typedef unsigned int u32x4 __attribute__((ext_vector_type(4)));
typedef int          i32x4 __attribute__((ext_vector_type(4)));
typedef int          i32x8 __attribute__((ext_vector_type(8)));

constexpr int KT  = 32;  // K tile (f16 WMMA 16x16x32)
constexpr int KTP = 40;  // padded LDS row stride in halves (80B = 5*16B => b128-aligned rows)

// ---------------- TDM (Tensor Data Mover) staging ----------------
// HBM -> LDS DMA without touching VGPRs: the CDNA5-preferred GEMM operand path.
#if defined(__HIP_DEVICE_COMPILE__) && defined(__gfx1250__) && \
    __has_builtin(__builtin_amdgcn_tensor_load_to_lds) &&      \
    __has_builtin(__builtin_amdgcn_s_wait_tensorcnt)
#define USE_TDM 1
#else
#define USE_TDM 0
#endif

#if USE_TDM
__device__ __forceinline__ i32x8 tdm_pack_g1(uint64_t q0, uint64_t q1, uint64_t q2,
                                             uint64_t q3) {
  union { uint64_t q[4]; i32x8 v; } u;
  u.q[0] = q0; u.q[1] = q1; u.q[2] = q2; u.q[3] = q3;
  return u.v;
}

// Load a 2D tile (rows x cols f16, row stride in halves) from global into LDS at
// byte offset lds_off, with optional LDS row padding (pad codes per D# group1).
// pad_interval_code: DWORDs before pad = 2<<code ; pad_amount_code: pad DWORDs = code+1.
__device__ __forceinline__ void tdm_load_tile_f16(const _Float16* g, uint32_t lds_off,
                                                  uint32_t cols_halves, uint32_t rows,
                                                  uint32_t row_stride_halves,
                                                  uint32_t pad_en,
                                                  uint32_t pad_interval_code,
                                                  uint32_t pad_amount_code) {
  uint64_t ga = (uint64_t)(uintptr_t)g;
  u32x4 g0;
  g0.x = 1u;                                   // count=1 (valid user descriptor)
  g0.y = lds_off;                              // lds_addr (bytes)
  g0.z = (uint32_t)ga;                         // global_addr[31:0]
  g0.w = (uint32_t)(ga >> 32) | (2u << 30);    // global_addr[56:32] | type=2 ("image")

  // D# group1 (256b): mask=0 | data_size=1(2B) | pad ctl | dims/strides
  uint64_t q0 = (1ull << 16)                                   // data_size = 2 bytes
              | ((uint64_t)(pad_en & 1u) << 20)
              | ((uint64_t)(pad_interval_code & 7u) << 22)
              | ((uint64_t)(pad_amount_code & 127u) << 25)
              | ((uint64_t)(cols_halves & 0xFFFFu) << 48);     // tensor_dim0[15:0]
  uint64_t q1 = ((uint64_t)(cols_halves >> 16))                // tensor_dim0[31:16]
              | ((uint64_t)rows << 16)                         // tensor_dim1
              | ((uint64_t)(cols_halves & 0xFFFFu) << 48);     // tile_dim0
  uint64_t q2 = ((uint64_t)(rows & 0xFFFFu))                   // tile_dim1 (tile_dim2=0)
              | ((uint64_t)row_stride_halves << 32);           // tensor_dim0_stride[31:0]
  uint64_t q3 = 0;                                             // stride hi, dim1_stride

  i32x8 g1 = tdm_pack_g1(q0, q1, q2, q3);
  i32x4 z4 = {0, 0, 0, 0};
#if defined(__clang_major__) && (__clang_major__ >= 23)
  i32x8 z8 = {0, 0, 0, 0, 0, 0, 0, 0};
  __builtin_amdgcn_tensor_load_to_lds(g0, g1, z4, z4, z8, 0);
#else
  __builtin_amdgcn_tensor_load_to_lds(g0, g1, z4, z4, 0);
#endif
}

__device__ __forceinline__ uint32_t lds_offset_of(const void* p) {
  // generic LDS pointer: low 32 bits carry the workgroup-relative LDS offset
  return (uint32_t)(uintptr_t)p;
}
#endif  // USE_TDM

// Build a 16-half WMMA fragment from LDS: per CDNA5 ISA, lane's 16 elements are
// two contiguous 8-element runs at K = khalf*8 and K = 16 + khalf*8.
__device__ __forceinline__ v16h ldfrag(const _Float16* __restrict__ p, int khalf) {
  v8h lo = *(const v8h*)(p + khalf * 8);
  v8h hi = *(const v8h*)(p + 16 + khalf * 8);
  v16h r;
#pragma unroll
  for (int i = 0; i < 8; ++i) { r[i] = lo[i]; r[i + 8] = hi[i]; }
  return r;
}

__device__ __forceinline__ float elu1(float x) {
  return x > 0.0f ? x + 1.0f : __expf(x);
}

// ---------------- f32 -> f16 convert ----------------
__global__ __launch_bounds__(256) void cvt_f32_f16(const float* __restrict__ in,
                                                   _Float16* __restrict__ out, int n) {
  int i = blockIdx.x * 256 + threadIdx.x;
  if (i < n) out[i] = (_Float16)in[i];
}

// ---------------- generic GEMM: C = A(MxK) @ B(NxK)^T ----------------
// EP: 0 = store f16, 1 = elu+1 then f16, 2 = store f32
// TDM double-buffered pipeline: DMA of tile i+1 overlaps WMMA on tile i
// (TENSORcnt completes in-order per wave, so s_wait_tensorcnt 2 == "previous
// A/B pair landed" while the newest pair stays in flight).
template <int EP>
__global__ __launch_bounds__(256) void gemm_bt(const _Float16* __restrict__ A,
                                               const _Float16* __restrict__ B,
                                               void* __restrict__ Cout,
                                               int K, int lda, int ldb, int ldc) {
  __shared__ __align__(16) _Float16 As[2][128 * KTP];
  __shared__ __align__(16) _Float16 Bs[2][128 * KTP];

  const int tid  = threadIdx.x;
  const int wave = tid >> 5, lane = tid & 31;
  const int wm = wave >> 1, wn = wave & 1;       // 4x2 wave grid, wave tile 32x64
  const int khalf = lane >> 4, l15 = lane & 15;
  const int m0 = blockIdx.x * 128, n0 = blockIdx.y * 128;
  const int niter = K / KT;

  v8f acc[2][4];
#pragma unroll
  for (int tm = 0; tm < 2; ++tm)
#pragma unroll
    for (int tn = 0; tn < 4; ++tn)
#pragma unroll
      for (int j = 0; j < 8; ++j) acc[tm][tn][j] = 0.0f;

#if USE_TDM
  if (wave == 0) {  // prologue: fill buffer 0
    tdm_load_tile_f16(&A[(size_t)m0 * lda], lds_offset_of(As[0]),
                      KT, 128, (uint32_t)lda, 1u, 3u, 3u);
    tdm_load_tile_f16(&B[(size_t)n0 * ldb], lds_offset_of(Bs[0]),
                      KT, 128, (uint32_t)ldb, 1u, 3u, 3u);
  }
#endif

  for (int it = 0; it < niter; ++it) {
    const int kk = it * KT;
    const int cur = it & 1;
#if USE_TDM
    if (wave == 0) {
      if (it + 1 < niter) {
        // prefetch next pair into the other buffer, then wait for current pair
        tdm_load_tile_f16(&A[(size_t)m0 * lda + kk + KT], lds_offset_of(As[cur ^ 1]),
                          KT, 128, (uint32_t)lda, 1u, 3u, 3u);
        tdm_load_tile_f16(&B[(size_t)n0 * ldb + kk + KT], lds_offset_of(Bs[cur ^ 1]),
                          KT, 128, (uint32_t)ldb, 1u, 3u, 3u);
        __builtin_amdgcn_s_wait_tensorcnt(2);
      } else {
        __builtin_amdgcn_s_wait_tensorcnt(0);
      }
    }
    __syncthreads();
#else
    // fallback: stage via VGPRs, 2 b128 chunks/thread per tile
#pragma unroll
    for (int i = 0; i < 2; ++i) {
      int c = tid + i * 256;
      int r = c >> 2, ck = (c & 3) * 8;
      *(v8h*)&As[cur][r * KTP + ck] = *(const v8h*)&A[(size_t)(m0 + r) * lda + kk + ck];
      *(v8h*)&Bs[cur][r * KTP + ck] = *(const v8h*)&B[(size_t)(n0 + r) * ldb + kk + ck];
    }
    __syncthreads();
#endif

    v16h af[2], bf[4];
#pragma unroll
    for (int tm = 0; tm < 2; ++tm)
      af[tm] = ldfrag(&As[cur][(wm * 32 + tm * 16 + l15) * KTP], khalf);
#pragma unroll
    for (int tn = 0; tn < 4; ++tn)
      bf[tn] = ldfrag(&Bs[cur][(wn * 64 + tn * 16 + l15) * KTP], khalf);

#pragma unroll
    for (int tm = 0; tm < 2; ++tm)
#pragma unroll
      for (int tn = 0; tn < 4; ++tn)
        acc[tm][tn] = __builtin_amdgcn_wmma_f32_16x16x32_f16(
            false, af[tm], false, bf[tn], (short)0, acc[tm][tn], false, false);
    __syncthreads();
  }

  // epilogue: C layout — row = j + 8*khalf within 16x16 tile, col = lane&15
#pragma unroll
  for (int tm = 0; tm < 2; ++tm)
#pragma unroll
    for (int tn = 0; tn < 4; ++tn)
#pragma unroll
      for (int j = 0; j < 8; ++j) {
        size_t gr = (size_t)(m0 + wm * 32 + tm * 16 + khalf * 8 + j);
        size_t gc = (size_t)(n0 + wn * 64 + tn * 16 + l15);
        float x = acc[tm][tn][j];
        if (EP == 1) x = elu1(x);
        if (EP == 2)
          ((float*)Cout)[gr * ldc + gc] = x;
        else
          ((_Float16*)Cout)[gr * ldc + gc] = (_Float16)x;
      }
}

// ---------------- kv partial: per (b,h,chunk) kv[f,d] = sum_s kf[s,f]*v[s,d] ----------
// grid = (nBH, NCH). Also emits ksum partials. LDS holds transposed tiles so both
// WMMA operands read contiguous K(=s) runs.
__global__ __launch_bounds__(256) void kv_partial(const _Float16* __restrict__ kf,
                                                  const _Float16* __restrict__ v,
                                                  float* __restrict__ kvpart,
                                                  float* __restrict__ kspart) {
  __shared__ __align__(16) _Float16 AsT[nFD * KTP];  // [f][s] 20KB
  __shared__ __align__(16) _Float16 VsT[nHD * KTP];  // [d][s] 5KB

  const int bh = blockIdx.x, chunk = blockIdx.y;
  const int b = bh >> 4, h = bh & 15;
  const int s0 = chunk * SC;
  const int tid = threadIdx.x;
  const int wave = tid >> 5, lane = tid & 31;
  const int khalf = lane >> 4, l15 = lane & 15;

  v8f acc[2][4];
#pragma unroll
  for (int tm = 0; tm < 2; ++tm)
#pragma unroll
    for (int tn = 0; tn < 4; ++tn)
#pragma unroll
      for (int j = 0; j < 8; ++j) acc[tm][tn][j] = 0.0f;

  for (int ks = 0; ks < SC; ks += KT) {
    // stage kf tile transposed: 32 s x 256 f -> AsT[f][s]
#pragma unroll
    for (int i = 0; i < 4; ++i) {
      int c = tid + i * 256;
      int s = c >> 5, f0 = (c & 31) * 8;
      int sg = s0 + ks + s;
      v8h kv8 = *(const v8h*)&kf[(((size_t)b * nS + sg) * nH + h) * nFD + f0];
#pragma unroll
      for (int e = 0; e < 8; ++e) AsT[(f0 + e) * KTP + s] = kv8[e];
    }
    {  // stage v tile transposed: 32 s x 64 d -> VsT[d][s]
      int s = tid >> 3, d0 = (tid & 7) * 8;
      int sg = s0 + ks + s;
      v8h vv = *(const v8h*)&v[(((size_t)b * nS + sg) * nH + h) * nHD + d0];
#pragma unroll
      for (int e = 0; e < 8; ++e) VsT[(d0 + e) * KTP + s] = vv[e];
    }
    __syncthreads();

    v16h af[2], bf[4];
#pragma unroll
    for (int tm = 0; tm < 2; ++tm)
      af[tm] = ldfrag(&AsT[(wave * 32 + tm * 16 + l15) * KTP], khalf);
#pragma unroll
    for (int tn = 0; tn < 4; ++tn)
      bf[tn] = ldfrag(&VsT[(tn * 16 + l15) * KTP], khalf);
#pragma unroll
    for (int tm = 0; tm < 2; ++tm)
#pragma unroll
      for (int tn = 0; tn < 4; ++tn)
        acc[tm][tn] = __builtin_amdgcn_wmma_f32_16x16x32_f16(
            false, af[tm], false, bf[tn], (short)0, acc[tm][tn], false, false);
    __syncthreads();
  }

#pragma unroll
  for (int tm = 0; tm < 2; ++tm)
#pragma unroll
    for (int tn = 0; tn < 4; ++tn)
#pragma unroll
      for (int j = 0; j < 8; ++j) {
        int f = wave * 32 + tm * 16 + khalf * 8 + j;
        int d = tn * 16 + l15;
        kvpart[(size_t)chunk * (nBH * nFD * nHD) + (size_t)bh * nFD * nHD + f * nHD + d] =
            acc[tm][tn][j];
      }

  // ksum partial: thread owns f = tid
  float kss = 0.0f;
  for (int s = 0; s < SC; ++s)
    kss += (float)kf[(((size_t)b * nS + s0 + s) * nH + h) * nFD + tid];
  kspart[(size_t)chunk * (nBH * nFD) + bh * nFD + tid] = kss;
}

// ---------------- reduce NCH slices: dst[i] = sum_c src[c*n + i] ----------------
__global__ __launch_bounds__(256) void reduce_slices(const float* __restrict__ src,
                                                     float* __restrict__ dst, int n) {
  int i = blockIdx.x * 256 + threadIdx.x;
  if (i < n) {
    float s = 0.0f;
#pragma unroll
    for (int c = 0; c < NCH; ++c) s += src[(size_t)c * n + i];
    dst[i] = s;
  }
}

// ---------------- output contraction: out[s,d] = (qf @ kv) / (qf . ksum + 1e-6) ----
// grid = (nS/128, nBH). kv (256x64 f32) preloaded transposed as f16 in LDS.
// TDM double-buffers the 128x32 qf tiles.
__global__ __launch_bounds__(256) void attn_out(const _Float16* __restrict__ qf,
                                                const float* __restrict__ kv,
                                                const float* __restrict__ ksum,
                                                _Float16* __restrict__ attn) {
  __shared__ __align__(16) _Float16 Bt[nHD * nFD];      // [d][f] 32KB
  __shared__ __align__(16) _Float16 As[2][128 * KTP];   // 2 x 10KB
  __shared__ float ksl[nFD];
  __shared__ float norml[128];

  const int bh = blockIdx.y;
  const int b = bh >> 4, h = bh & 15;
  const int s0 = blockIdx.x * 128;
  const int tid = threadIdx.x;
  const int wave = tid >> 5, lane = tid & 31;
  const int khalf = lane >> 4, l15 = lane & 15;
  const size_t qrow0 = (((size_t)b * nS + s0) * nH + h) * nFD;  // row stride nH*nFD

  // preload kv transposed (f16) and ksum
#pragma unroll
  for (int i = 0; i < 64; ++i) {
    int idx = tid + i * 256;                // coalesced over f*64+d
    int f = idx >> 6, d = idx & 63;
    Bt[d * nFD + f] = (_Float16)kv[(size_t)bh * nFD * nHD + idx];
  }
  ksl[tid] = ksum[(size_t)bh * nFD + tid];

#if USE_TDM
  if (wave == 0) {  // prologue: first qf tile into buffer 0
    tdm_load_tile_f16(&qf[qrow0], lds_offset_of(As[0]),
                      KT, 128, (uint32_t)(nH * nFD), 1u, 3u, 3u);
  }
#endif
  __syncthreads();

  // per-row normalizer
  if (tid < 128) {
    int sg = s0 + tid;
    const _Float16* qp = &qf[(((size_t)b * nS + sg) * nH + h) * nFD];
    float nv = 0.0f;
#pragma unroll
    for (int c = 0; c < 32; ++c) {
      v8h q8 = *(const v8h*)(qp + c * 8);
#pragma unroll
      for (int e = 0; e < 8; ++e) nv += (float)q8[e] * ksl[c * 8 + e];
    }
    norml[tid] = nv + 1e-6f;
  }
  __syncthreads();

  v8f acc[4];
#pragma unroll
  for (int tn = 0; tn < 4; ++tn)
#pragma unroll
    for (int j = 0; j < 8; ++j) acc[tn][j] = 0.0f;

  constexpr int niter = nFD / KT;  // 8
  for (int it = 0; it < niter; ++it) {
    const int kk = it * KT;
    const int cur = it & 1;
#if USE_TDM
    if (wave == 0) {
      if (it + 1 < niter) {
        tdm_load_tile_f16(&qf[qrow0 + kk + KT], lds_offset_of(As[cur ^ 1]),
                          KT, 128, (uint32_t)(nH * nFD), 1u, 3u, 3u);
        __builtin_amdgcn_s_wait_tensorcnt(1);
      } else {
        __builtin_amdgcn_s_wait_tensorcnt(0);
      }
    }
    __syncthreads();
#else
    // stage qf tile 128 x 32 (contiguous f)
#pragma unroll
    for (int i = 0; i < 2; ++i) {
      int c = tid + i * 256;
      int r = c >> 2, ck = (c & 3) * 8;
      *(v8h*)&As[cur][r * KTP + ck] =
          *(const v8h*)&qf[(((size_t)b * nS + s0 + r) * nH + h) * nFD + kk + ck];
    }
    __syncthreads();
#endif

    v16h af = ldfrag(&As[cur][(wave * 16 + l15) * KTP], khalf);
#pragma unroll
    for (int tn = 0; tn < 4; ++tn) {
      v16h bf = ldfrag(&Bt[(tn * 16 + l15) * nFD + kk], khalf);
      acc[tn] = __builtin_amdgcn_wmma_f32_16x16x32_f16(
          false, af, false, bf, (short)0, acc[tn], false, false);
    }
    __syncthreads();
  }

#pragma unroll
  for (int tn = 0; tn < 4; ++tn)
#pragma unroll
    for (int j = 0; j < 8; ++j) {
      int r = wave * 16 + khalf * 8 + j;
      int d = tn * 16 + l15;
      float val = acc[tn][j] / norml[r];
      attn[(((size_t)b * nS + s0 + r) * nH + h) * nHD + d] = (_Float16)val;
    }
}

// ---------------- launch ----------------
extern "C" void kernel_launch(void* const* d_in, const int* in_sizes, int n_in,
                              void* d_out, int out_size, void* d_ws, size_t ws_size,
                              hipStream_t stream) {
  const float* hs = (const float*)d_in[0];
  const float* Wq = (const float*)d_in[1];
  const float* Wk = (const float*)d_in[2];
  const float* Wv = (const float*)d_in[3];
  const float* Wo = (const float*)d_in[4];
  const float* Wf = (const float*)d_in[5];
  float* out = (float*)d_out;

  // workspace layout (bytes, 256-aligned)
  constexpr size_t AL = 256;
  auto rnd = [](size_t x) { return (x + AL - 1) & ~(AL - 1); };
  constexpr size_t nHS = (size_t)nB * nS * nD;          // 16.7M
  constexpr size_t nW  = (size_t)nD * nD;
  constexpr size_t nWf = (size_t)nFD * nHD;
  constexpr size_t nFt = (size_t)nB * nS * nH * nFD;    // 67.1M
  constexpr size_t nKV = (size_t)nBH * nFD * nHD;       // 524288
  constexpr size_t nKS = (size_t)nBH * nFD;             // 8192

  size_t off = 0;
  char* ws = (char*)d_ws;
  auto take = [&](size_t bytes) { char* p = ws + off; off += rnd(bytes); return p; };

  _Float16* hsH = (_Float16*)take(nHS * 2);
  _Float16* wqH = (_Float16*)take(nW * 2);
  _Float16* wkH = (_Float16*)take(nW * 2);
  _Float16* wvH = (_Float16*)take(nW * 2);
  _Float16* woH = (_Float16*)take(nW * 2);
  _Float16* wfH = (_Float16*)take(nWf * 2);
  _Float16* qH  = (_Float16*)take(nHS * 2);
  _Float16* kH  = (_Float16*)take(nHS * 2);
  _Float16* vH  = (_Float16*)take(nHS * 2);
  _Float16* qfH = (_Float16*)take(nFt * 2);
  _Float16* kfH = (_Float16*)take(nFt * 2);
  float* kspart = (float*)take(nKS * NCH * 4);
  float* kvF    = (float*)take(nKV * 4);
  float* ksF    = (float*)take(nKS * 4);
  // aliases (lifetimes are disjoint): kvpart over kH (same 32MiB), attn over qH
  float* kvpart   = (float*)kH;
  _Float16* attnH = qH;

  // 1) convert to f16
  cvt_f32_f16<<<(int)(nHS / 256), 256, 0, stream>>>(hs, hsH, (int)nHS);
  cvt_f32_f16<<<(int)(nW / 256), 256, 0, stream>>>(Wq, wqH, (int)nW);
  cvt_f32_f16<<<(int)(nW / 256), 256, 0, stream>>>(Wk, wkH, (int)nW);
  cvt_f32_f16<<<(int)(nW / 256), 256, 0, stream>>>(Wv, wvH, (int)nW);
  cvt_f32_f16<<<(int)(nW / 256), 256, 0, stream>>>(Wo, woH, (int)nW);
  cvt_f32_f16<<<(int)(nWf / 256), 256, 0, stream>>>(Wf, wfH, (int)nWf);

  // 2) q/k/v projections: M = nB*nS = 16384, N = K = 1024
  {
    dim3 g(nB * nS / 128, nD / 128);
    gemm_bt<0><<<g, 256, 0, stream>>>(hsH, wqH, qH, nD, nD, nD, nD);
    gemm_bt<0><<<g, 256, 0, stream>>>(hsH, wkH, kH, nD, nD, nD, nD);
    gemm_bt<0><<<g, 256, 0, stream>>>(hsH, wvH, vH, nD, nD, nD, nD);
  }

  // 3) feature maps: M = nB*nS*nH = 262144, N = 256, K = 64, fused elu+1
  {
    dim3 g(nB * nS * nH / 128, nFD / 128);
    gemm_bt<1><<<g, 256, 0, stream>>>(qH, wfH, qfH, nHD, nHD, nHD, nFD);
    gemm_bt<1><<<g, 256, 0, stream>>>(kH, wfH, kfH, nHD, nHD, nHD, nFD);
  }

  // 4) kv + ksum partials over S slices
  {
    dim3 g(nBH, NCH);
    kv_partial<<<g, 256, 0, stream>>>(kfH, vH, kvpart, kspart);
  }

  // 5) reduce slices
  reduce_slices<<<(int)(nKV / 256), 256, 0, stream>>>(kvpart, kvF, (int)nKV);
  reduce_slices<<<(int)(nKS / 256), 256, 0, stream>>>(kspart, ksF, (int)nKS);

  // 6) normalized output contraction -> attn (f16, layout B,S,H,HD == B,S,D)
  {
    dim3 g(nS / 128, nBH);
    attn_out<<<g, 256, 0, stream>>>(qfH, kvF, ksF, attnH);
  }

  // 7) final projection: out = attn @ Wo^T (f32 output)
  {
    dim3 g(nB * nS / 128, nD / 128);
    gemm_bt<2><<<g, 256, 0, stream>>>(attnH, woH, out, nD, nD, nD, nD);
  }

  (void)in_sizes; (void)n_in; (void)out_size; (void)ws_size;
}
